// CausalSelfAttention_15848429322879
// MI455X (gfx1250) — matrix operand order repaired and checked by reading, not compiled, not run on hardware
//
#include <hip/hip_runtime.h>
#include <hip/hip_bf16.h>
#include <stdint.h>

// ---------------------------------------------------------------------------
// Causal self-attention for B=2, T=2048, C=1024, H=16, D=64 on gfx1250.
// Pipeline: [x @ Wqkv + b -> bf16 qkv] -> [flash attention -> bf16 y]
//           -> [y @ Wproj + b -> f32 out], all matmuls on v_wmma_f32_16x16x32_bf16.
// Tile staging uses GLOBAL_LOAD_ASYNC_TO_LDS (ASYNCcnt) where the copy is
// layout-preserving; transposing/converting stages remain VGPR-mediated.
// ---------------------------------------------------------------------------

typedef __attribute__((ext_vector_type(16))) __bf16 v16bf;
typedef __attribute__((ext_vector_type(8)))  float  v8f;
typedef __attribute__((ext_vector_type(4)))  int    v4i;

#define B_  2
#define T_  2048
#define C_  1024
#define H_  16
#define D_  64
#define C3_ 3072

#if defined(__gfx1250__) && __has_builtin(__builtin_amdgcn_global_load_async_to_lds_b128)
#define HAVE_ASYNC_LDS 1
#else
#define HAVE_ASYNC_LDS 0
#endif

#if HAVE_ASYNC_LDS
// Builtin signature (from clang diagnostic): param0 = v4i in addrspace(1) [__device__],
// param1 = v4i in addrspace(3) [LDS], then imm offset, imm cpol.
typedef __attribute__((address_space(1))) v4i* gbl_v4i;
typedef __attribute__((address_space(3))) v4i* lds_v4i;
static __device__ __forceinline__ lds_v4i to_lds(void* p) {
  // generic LDS pointer: low 32 bits == LDS offset (flat aperture rule)
  return (lds_v4i)(unsigned int)(uintptr_t)p;
}
static __device__ __forceinline__ gbl_v4i to_gbl(const void* p) {
  return (gbl_v4i)(uintptr_t)p;
}
static __device__ __forceinline__ void wait_async0() {
#if __has_builtin(__builtin_amdgcn_s_wait_asynccnt)
  __builtin_amdgcn_s_wait_asynccnt(0);
#else
  asm volatile("s_wait_asynccnt 0" ::: "memory");
#endif
}
#endif

static __device__ __forceinline__ unsigned short f2bf(float f) {
  union { float f; unsigned u; } x; x.f = f;
  unsigned r = x.u + 0x7FFFu + ((x.u >> 16) & 1u);   // round-to-nearest-even
  return (unsigned short)(r >> 16);
}

union FragAB { v16bf v; unsigned u[8]; };

static __device__ __forceinline__ v8f wmma_bf16(v16bf a, v16bf b, v8f c) {
  return __builtin_amdgcn_wmma_f32_16x16x32_bf16(
      /*neg_a=*/false, a, /*neg_b=*/false, b,
      /*c_mod=*/(short)0, c, /*reuse_a=*/false, /*reuse_b=*/false);
}

// ---------------------------------------------------------------------------
// Generic bf16 WMMA GEMM: C[M,N] = A[M,K] @ W[K,N] + bias[N]
// Block tile 128x128, BK=32, 256 threads (8 waves), wave tile 64x32 (4x2 WMMAs).
// ---------------------------------------------------------------------------
template<bool A_BF16, bool OUT_BF16>
__global__ __launch_bounds__(256)
void gemm_wmma(const void* __restrict__ Ap, const float* __restrict__ W,
               const float* __restrict__ bias, void* __restrict__ Cp,
               int N, int K)
{
  __shared__ unsigned short As[128][40];   // [m][k], 80B rows (16-aligned)
  __shared__ unsigned short Bs[128][36];   // [n][k], padded (W tile transposed)

  const int tid  = threadIdx.x;
  const int lane = tid & 31, wid = tid >> 5;
  const int r  = lane & 15, hh = lane >> 4;
  const int wm = wid >> 2,  wn = wid & 3;
  const int rowBase = blockIdx.y * 128;
  const int colBase = blockIdx.x * 128;

  v8f acc[4][2];
  #pragma unroll
  for (int i = 0; i < 4; ++i)
    #pragma unroll
    for (int j = 0; j < 2; ++j) acc[i][j] = (v8f){0,0,0,0,0,0,0,0};

  for (int kb = 0; kb < K; kb += 32) {
    __syncthreads();
    // ---- Stage A tile 128x32 ----
#if HAVE_ASYNC_LDS
    if (A_BF16) {
      // direct async copy, one B128 per thread x2 (8 KB tile)
      #pragma unroll
      for (int it = 0; it < 2; ++it) {
        int idx = it * 256 + tid;
        int row = idx >> 2, seg = idx & 3;      // 4 x 16B segments per 64B row
        const unsigned short* src =
            (const unsigned short*)Ap + (size_t)(rowBase + row) * K + kb + seg * 8;
        __builtin_amdgcn_global_load_async_to_lds_b128(
            to_gbl(src), to_lds(&As[row][seg * 8]), 0, 0);
      }
    } else
#endif
    {
      #pragma unroll
      for (int it = 0; it < 16; ++it) {
        int idx = it * 256 + tid;
        int row = idx >> 5, col = idx & 31;
        unsigned short v;
        if (A_BF16) v = ((const unsigned short*)Ap)[(size_t)(rowBase + row) * K + kb + col];
        else        v = f2bf(((const float*)Ap)[(size_t)(rowBase + row) * K + kb + col]);
        As[row][col] = v;
      }
    }
    // ---- Stage W tile 32x128 transposed into Bs[n][k] (f32 -> bf16) ----
    #pragma unroll
    for (int it = 0; it < 16; ++it) {
      int idx = it * 256 + tid;
      int krow = idx >> 7, ncol = idx & 127;
      Bs[ncol][krow] = f2bf(W[(size_t)(kb + krow) * N + colBase + ncol]);
    }
#if HAVE_ASYNC_LDS
    if (A_BF16) wait_async0();
#endif
    __syncthreads();

    // Prefetch next K-step tiles into cache while this step computes.
    if (kb + 32 < K) {
      size_t aoff = (size_t)(rowBase + (tid >> 1)) * K + (kb + 32) + (tid & 1) * 16;
      if (A_BF16) __builtin_prefetch((const unsigned short*)Ap + aoff, 0, 0);
      else        __builtin_prefetch((const float*)Ap + aoff, 0, 0);
      __builtin_prefetch(W + (size_t)(kb + 32 + (tid >> 7)) * N + colBase + (tid & 127), 0, 0);
    }

    FragAB a[4], b[2];
    #pragma unroll
    for (int mt = 0; mt < 4; ++mt) {
      const unsigned short* pr = &As[wm * 64 + mt * 16 + r][0];
      #pragma unroll
      for (int j = 0; j < 8; ++j) {
        int k0 = (j < 4) ? (8 * hh + 2 * j) : (16 + 8 * hh + 2 * (j - 4));
        a[mt].u[j] = *(const unsigned int*)(pr + k0);
      }
    }
    #pragma unroll
    for (int nt = 0; nt < 2; ++nt) {
      const unsigned short* pc = &Bs[wn * 32 + nt * 16 + r][0];
      #pragma unroll
      for (int j = 0; j < 8; ++j) {
        int k0 = (j < 4) ? (8 * hh + 2 * j) : (16 + 8 * hh + 2 * (j - 4));
        b[nt].u[j] = *(const unsigned int*)(pc + k0);
      }
    }
    #pragma unroll
    for (int mt = 0; mt < 4; ++mt)
      #pragma unroll
      for (int nt = 0; nt < 2; ++nt)
        acc[mt][nt] = wmma_bf16(a[mt].v, b[nt].v, acc[mt][nt]);
  }

  // Epilogue: C layout -> row = v + 8*half, col = lane&15
  #pragma unroll
  for (int mt = 0; mt < 4; ++mt)
    #pragma unroll
    for (int nt = 0; nt < 2; ++nt)
      #pragma unroll
      for (int v = 0; v < 8; ++v) {
        int row = rowBase + wm * 64 + mt * 16 + v + 8 * hh;
        int col = colBase + wn * 32 + nt * 16 + r;
        float val = acc[mt][nt][v] + bias[col];
        if (OUT_BF16) ((unsigned short*)Cp)[(size_t)row * N + col] = f2bf(val);
        else          ((float*)Cp)[(size_t)row * N + col] = val;
      }
}

// ---------------------------------------------------------------------------
// Flash attention: block = (q-tile of 128, head, batch); wave owns 16 q rows.
// Streams 32-key tiles: S = Q K^T (WMMA), causal mask, online softmax,
// O += P V (WMMA).  All operands bf16, f32 accumulate.
// ---------------------------------------------------------------------------
__global__ __launch_bounds__(256)
void flash_attn(const unsigned short* __restrict__ qkv, unsigned short* __restrict__ y)
{
  __shared__ unsigned short Ks[32][72];      // [key][d], 144B rows (16-aligned)
  __shared__ unsigned short Vt[64][34];      // [d][key] (transposed)
  __shared__ unsigned short Pw[8][16][36];   // per-wave P re-layout buffer

  const int tid  = threadIdx.x;
  const int lane = tid & 31, wid = tid >> 5;
  const int r  = lane & 15, hh = lane >> 4;
  const int qt = blockIdx.x, h = blockIdx.y, b = blockIdx.z;
  const int q0 = qt * 128;
  const int qw = q0 + wid * 16;

  // Q fragments (A layout), D=64 split as two 32-wide K chunks
  FragAB qf[2];
  {
    const unsigned short* qrow = qkv + (size_t)(b * T_ + qw + r) * C3_ + h * D_;
    #pragma unroll
    for (int j = 0; j < 8; ++j) {
      int k0 = (j < 4) ? (8 * hh + 2 * j) : (16 + 8 * hh + 2 * (j - 4));
      qf[0].u[j] = *(const unsigned int*)(qrow + k0);
      qf[1].u[j] = *(const unsigned int*)(qrow + 32 + k0);
    }
  }

  float mrow[8], lrow[8];
  #pragma unroll
  for (int v = 0; v < 8; ++v) { mrow[v] = -3.0e38f; lrow[v] = 0.0f; }
  v8f o[4];
  #pragma unroll
  for (int t = 0; t < 4; ++t) o[t] = (v8f){0,0,0,0,0,0,0,0};

  const int nkt = (q0 + 128) / 32;   // causal key-tile bound for this block
  const float scale = 0.125f;        // 1/sqrt(64)

  for (int kt = 0; kt < nkt; ++kt) {
    const int kbase = kt * 32;
    __syncthreads();
    // ---- Stage K tile (32 keys x 64 d) ----
#if HAVE_ASYNC_LDS
    {
      int key = tid >> 3, seg = tid & 7;     // 8 x 16B segments per 128B key row
      const unsigned short* src =
          qkv + (size_t)(b * T_ + kbase + key) * C3_ + C_ + h * D_ + seg * 8;
      __builtin_amdgcn_global_load_async_to_lds_b128(
          to_gbl(src), to_lds(&Ks[key][seg * 8]), 0, 0);
    }
#else
    #pragma unroll
    for (int it = 0; it < 4; ++it) {
      int idx = it * 256 + tid;
      int key = idx >> 5, dw = idx & 31;
      const unsigned int* src = (const unsigned int*)
          (qkv + (size_t)(b * T_ + kbase + key) * C3_ + C_ + h * D_ + dw * 2);
      *(unsigned int*)(&Ks[key][dw * 2]) = *src;
    }
#endif
    // ---- Stage V tile transposed: Vt[d][key] (layout change -> VGPR path) ----
    #pragma unroll
    for (int it = 0; it < 8; ++it) {
      int idx = it * 256 + tid;
      int key = idx >> 6, d = idx & 63;
      Vt[d][key] = qkv[(size_t)(b * T_ + kbase + key) * C3_ + 2 * C_ + h * D_ + d];
    }
#if HAVE_ASYNC_LDS
    wait_async0();
#endif
    __syncthreads();

    // S = Q @ K^T : two 16-key subtiles, chained over D
    v8f s[2];
    #pragma unroll
    for (int nt = 0; nt < 2; ++nt) {
      FragAB bk0, bk1;
      #pragma unroll
      for (int j = 0; j < 8; ++j) {
        int k0 = (j < 4) ? (8 * hh + 2 * j) : (16 + 8 * hh + 2 * (j - 4));
        bk0.u[j] = *(const unsigned int*)(&Ks[nt * 16 + r][k0]);
        bk1.u[j] = *(const unsigned int*)(&Ks[nt * 16 + r][32 + k0]);
      }
      v8f z = (v8f){0,0,0,0,0,0,0,0};
      z = wmma_bf16(qf[0].v, bk0.v, z);
      z = wmma_bf16(qf[1].v, bk1.v, z);
      s[nt] = z;
    }
    // Scale + causal mask
    #pragma unroll
    for (int nt = 0; nt < 2; ++nt)
      #pragma unroll
      for (int v = 0; v < 8; ++v) {
        int qi = qw + v + 8 * hh;
        int kj = kbase + nt * 16 + r;
        float val = s[nt][v] * scale;
        s[nt][v] = (kj > qi) ? -1.0e30f : val;
      }
    // Online softmax (row spans 16 lanes of a half -> xor-butterfly over 1,2,4,8)
    #pragma unroll
    for (int v = 0; v < 8; ++v) {
      float tmax = fmaxf(s[0][v], s[1][v]);
      #pragma unroll
      for (int off = 1; off < 16; off <<= 1)
        tmax = fmaxf(tmax, __shfl_xor(tmax, off, 32));
      float mnew  = fmaxf(mrow[v], tmax);
      float alpha = __expf(mrow[v] - mnew);
      mrow[v] = mnew;
      float p0 = __expf(s[0][v] - mnew);
      float p1 = __expf(s[1][v] - mnew);
      s[0][v] = p0; s[1][v] = p1;
      float rs = p0 + p1;
      #pragma unroll
      for (int off = 1; off < 16; off <<= 1)
        rs += __shfl_xor(rs, off, 32);
      lrow[v] = lrow[v] * alpha + rs;
      #pragma unroll
      for (int t = 0; t < 4; ++t) o[t][v] *= alpha;
    }
    // C-layout -> A-layout for P via per-wave LDS
    #pragma unroll
    for (int nt = 0; nt < 2; ++nt)
      #pragma unroll
      for (int v = 0; v < 8; ++v)
        Pw[wid][v + 8 * hh][nt * 16 + r] = f2bf(s[nt][v]);
    asm volatile("s_wait_dscnt 0" ::: "memory");   // wave-local LDS RAW
    FragAB pf;
    #pragma unroll
    for (int j = 0; j < 8; ++j) {
      int k0 = (j < 4) ? (8 * hh + 2 * j) : (16 + 8 * hh + 2 * (j - 4));
      pf.u[j] = *(const unsigned int*)(&Pw[wid][r][k0]);
    }
    // O += P @ V over four 16-wide d slices
    #pragma unroll
    for (int dt = 0; dt < 4; ++dt) {
      FragAB vf;
      #pragma unroll
      for (int j = 0; j < 8; ++j)
        vf.u[j] = *(const unsigned int*)(&Vt[dt * 16 + r][(j < 4) ? (8 * hh + 2 * j) : (16 + 8 * hh + 2 * (j - 4))]);
      o[dt] = wmma_bf16(pf.v, vf.v, o[dt]);
    }
  }

  // Normalize and emit y (bf16)
  #pragma unroll
  for (int dt = 0; dt < 4; ++dt)
    #pragma unroll
    for (int v = 0; v < 8; ++v) {
      int qi = qw + v + 8 * hh;
      float val = o[dt][v] / lrow[v];
      y[(size_t)(b * T_ + qi) * C_ + h * D_ + dt * 16 + r] = f2bf(val);
    }
}

// ---------------------------------------------------------------------------
extern "C" void kernel_launch(void* const* d_in, const int* in_sizes, int n_in,
                              void* d_out, int out_size, void* d_ws, size_t ws_size,
                              hipStream_t stream)
{
  const float* x     = (const float*)d_in[0];
  const float* Wqkv  = (const float*)d_in[1];
  const float* bqkv  = (const float*)d_in[2];
  const float* Wproj = (const float*)d_in[3];
  const float* bproj = (const float*)d_in[4];

  unsigned short* qkv = (unsigned short*)d_ws;                 // [4096][3072] bf16
  unsigned short* yb  = qkv + (size_t)B_ * T_ * C3_;           // [4096][1024] bf16

  dim3 blk(256);
  // 1) qkv = x @ Wqkv + bqkv  (f32 in, bf16 out)
  gemm_wmma<false, true><<<dim3(C3_ / 128, (B_ * T_) / 128), blk, 0, stream>>>(
      x, Wqkv, bqkv, qkv, C3_, C_);
  // 2) flash attention per (q-tile, head, batch)
  flash_attn<<<dim3(T_ / 128, H_, B_), blk, 0, stream>>>(qkv, yb);
  // 3) out = y @ Wproj + bproj (bf16 in, f32 out)
  gemm_wmma<true, false><<<dim3(C_ / 128, (B_ * T_) / 128), blk, 0, stream>>>(
      yb, Wproj, bproj, d_out, C_, C_);
}
